// Model_34454227648736
// MI455X (gfx1250) — matrix operand order repaired and checked by reading, not compiled, not run on hardware
//
#include <hip/hip_runtime.h>
#include <hip/hip_bf16.h>
#include <math.h>

// ---------------------------------------------------------------------------
// ViT-S/16 (shifted-patch) forward for MI455X (gfx1250, wave32, WMMA)
// ---------------------------------------------------------------------------
#define BB    64
#define IMGSZ 224
#define PSZ   16
#define GRD   14
#define NPAT  196
#define NTOK  197
#define DIMD  384
#define DEPTH 12
#define NH    12
#define DHD   32
#define MLPD  1536
#define NCLS  1000
#define PDIM  1792          // 7*16*16
#define SEQP  208           // 197 padded to 16
#define SEQK  224           // 197 padded to 32 (K dim of P@V)

typedef __attribute__((ext_vector_type(16))) _Float16 v16h;
typedef __attribute__((ext_vector_type(8)))  _Float16 v8h;
typedef __attribute__((ext_vector_type(8)))  float    v8f;

union Frag { v16h v; v8h h[2]; };

// Force the GLOBAL address space (AS1) so loads lower to global_load_b128
// (LOADcnt only) instead of flat_load_b128 (LOADcnt + DScnt + aperture check).
#define GAS __attribute__((address_space(1)))

static __device__ __forceinline__ v8h gload8(const _Float16* p) {
  return *(const GAS v8h*)((const GAS _Float16*)p);
}
static __device__ __forceinline__ float gloadf(const float* p) {
  return *(const GAS float*)p;
}
static __device__ __forceinline__ void gstoref(float* p, float v) {
  *(GAS float*)p = v;
}
static __device__ __forceinline__ void gstoreh(_Float16* p, _Float16 v) {
  *(GAS _Float16*)p = v;
}

// ---------------------------------------------------------------------------
// Generic f32 -> f16 conversion
// ---------------------------------------------------------------------------
__global__ void f32_to_f16_kernel(const float* __restrict__ in,
                                  _Float16* __restrict__ out, int n) {
  int i = blockIdx.x * blockDim.x + threadIdx.x;
  if (i < n) out[i] = (_Float16)in[i];
}

// ---------------------------------------------------------------------------
// Build shifted-patch A matrix: (B*196) x 1792, f16.
// pd = (py*16+px)*7 + c ; channels 0..2 = img, 3..6 = shifted channel means.
// ---------------------------------------------------------------------------
__global__ void build_patches_kernel(const float* __restrict__ img,
                                     _Float16* __restrict__ Ap, int total) {
  int idx = blockIdx.x * blockDim.x + threadIdx.x;
  if (idx >= total) return;
  int pd = idx % PDIM;
  int r  = idx / PDIM;
  int b = r / NPAT, patch = r % NPAT;
  int gy = patch / GRD, gx = patch % GRD;
  int c = pd % 7, pp = pd / 7;
  int py = pp / PSZ, px = pp % PSZ;
  int y = gy * PSZ + py, x = gx * PSZ + px;
  const float* ib = img + (size_t)b * 3 * IMGSZ * IMGSZ;
  float val;
  if (c < 3) {
    val = ib[(c * IMGSZ + y) * IMGSZ + x];
  } else {
    int yy = y, xx = x;
    if      (c == 3) xx = x - 8;   // left-shift view
    else if (c == 4) xx = x + 8;   // right
    else if (c == 5) yy = y - 8;   // top
    else             yy = y + 8;   // bottom
    if (yy < 0 || yy >= IMGSZ || xx < 0 || xx >= IMGSZ) {
      val = 0.f;
    } else {
      val = (ib[(0 * IMGSZ + yy) * IMGSZ + xx] +
             ib[(1 * IMGSZ + yy) * IMGSZ + xx] +
             ib[(2 * IMGSZ + yy) * IMGSZ + xx]) * (1.f / 3.f);
    }
  }
  Ap[idx] = (_Float16)val;
}

// ---------------------------------------------------------------------------
// Assemble x = concat(cls, patches) + pos_emb   -> (B,197,384) f32
// ---------------------------------------------------------------------------
__global__ void assemble_x_kernel(const float* __restrict__ xpatch,
                                  const float* __restrict__ cls_tok,
                                  const float* __restrict__ pos_emb,
                                  float* __restrict__ x, int total) {
  int idx = blockIdx.x * blockDim.x + threadIdx.x;
  if (idx >= total) return;
  int d = idx % DIMD;
  int t = (idx / DIMD) % NTOK;
  int b = idx / (DIMD * NTOK);
  float v = (t == 0) ? cls_tok[d]
                     : xpatch[((size_t)b * NPAT + (t - 1)) * DIMD + d];
  x[idx] = v + pos_emb[t * DIMD + d];
}

// ---------------------------------------------------------------------------
// LayerNorm over 384 elements, one row per block (128 threads, wave32
// shuffle reduce + cross-wave LDS).  Emits f16 (next GEMM's A operand).
// ---------------------------------------------------------------------------
__global__ __launch_bounds__(128) void layernorm_kernel(
    const float* __restrict__ in, size_t rowPitch,
    const float* __restrict__ g, const float* __restrict__ bta,
    _Float16* __restrict__ out) {
  const int row = blockIdx.x;
  const float* p = in + (size_t)row * rowPitch;
  const int t = threadIdx.x;
  float x0 = p[t], x1 = p[t + 128], x2 = p[t + 256];
  float s  = x0 + x1 + x2;
  float ss = x0 * x0 + x1 * x1 + x2 * x2;
  #pragma unroll
  for (int o = 16; o > 0; o >>= 1) {
    s  += __shfl_xor(s,  o, 32);
    ss += __shfl_xor(ss, o, 32);
  }
  __shared__ float rs[4], rss[4];
  __shared__ float mu_s, rsig_s;
  int w = t >> 5;
  if ((t & 31) == 0) { rs[w] = s; rss[w] = ss; }
  __syncthreads();
  if (t == 0) {
    float S = rs[0] + rs[1] + rs[2] + rs[3];
    float SS = rss[0] + rss[1] + rss[2] + rss[3];
    float mu = S * (1.f / DIMD);
    float var = SS * (1.f / DIMD) - mu * mu;
    mu_s = mu;
    rsig_s = rsqrtf(var + 1e-5f);
  }
  __syncthreads();
  float mu = mu_s, rsig = rsig_s;
  out[(size_t)row * DIMD + t]       = (_Float16)((x0 - mu) * rsig * g[t]       + bta[t]);
  out[(size_t)row * DIMD + t + 128] = (_Float16)((x1 - mu) * rsig * g[t + 128] + bta[t + 128]);
  out[(size_t)row * DIMD + t + 256] = (_Float16)((x2 - mu) * rsig * g[t + 256] + bta[t + 256]);
}

// ---------------------------------------------------------------------------
// Split qkv f32 (B,197,1152) -> q,k f16 (B,H,208,32) zero-padded rows and
// v transposed f16 (B,H,32,224) zero-padded cols (K dim for P@V WMMA).
// ---------------------------------------------------------------------------
__global__ void qkv_split_kernel(const float* __restrict__ qkv,
                                 _Float16* __restrict__ qh,
                                 _Float16* __restrict__ kh,
                                 _Float16* __restrict__ vt, int total) {
  int idx = blockIdx.x * blockDim.x + threadIdx.x;
  if (idx >= total) return;                // total = B*H*SEQK*32
  int d = idx % DHD;
  int j = (idx / DHD) % SEQK;
  int h = (idx / (DHD * SEQK)) % NH;
  int b = idx / (DHD * SEQK * NH);
  float q = 0.f, k = 0.f, v = 0.f;
  if (j < NTOK) {
    const float* p = qkv + ((size_t)b * NTOK + j) * (3 * DIMD) + h * DHD + d;
    q = p[0]; k = p[DIMD]; v = p[2 * DIMD];
  }
  size_t bh = (size_t)b * NH + h;
  if (j < SEQP) {
    qh[(bh * SEQP + j) * DHD + d] = (_Float16)q;
    kh[(bh * SEQP + j) * DHD + d] = (_Float16)k;
  }
  vt[(bh * DHD + d) * SEQK + j] = (_Float16)v;
}

// ---------------------------------------------------------------------------
// WMMA GEMM:  C(MxN) = act( A(MxK,f16) @ W(NxK,f16)^T + bias [+ R] )
// Block = 8 waves (4 x 2), wave tile 32x64 (2x4 WMMA 16x16x32 tiles),
// block tile 128x128.  Loads are divergence-free: out-of-range row/col
// indices are CLAMPED (valid address, garbage data) and discarded by the
// epilogue store guards.  8 WMMAs per 6 fragment loads per K-step.
// act=1 applies exact GELU. C (f32) and Ch (f16) stores are each optional.
// A-frag: lane row=l&15, halves {h*8..+7, 16+h*8..+7}  (ISA 16-bit A layout)
// B-frag: lane col=l&15, halves {h*8..+7, 16+h*8..+7}  (matches A layout)
// ---------------------------------------------------------------------------
__global__ __launch_bounds__(256) void gemm_wmma_kernel(
    const _Float16* __restrict__ A, const _Float16* __restrict__ W,
    const float* __restrict__ bias, const float* __restrict__ R,
    float* __restrict__ C, _Float16* __restrict__ Ch,
    int M, int N, int K, int act) {
  const int lane = threadIdx.x & 31;
  const int wave = threadIdx.x >> 5;
  const int wm = wave & 3, wn = wave >> 2;
  const int row0 = blockIdx.x * 128 + wm * 32;
  const int col0 = blockIdx.y * 128 + wn * 64;
  const int ll = lane & 15, hh = lane >> 4;

  // Divergence-free fragment base pointers (clamped indices).
  const _Float16* pa[2];
  #pragma unroll
  for (int mi = 0; mi < 2; ++mi) {
    int r = min(row0 + mi * 16 + ll, M - 1);
    pa[mi] = A + (size_t)r * K + hh * 8;
  }
  const _Float16* pb[4];
  #pragma unroll
  for (int ni = 0; ni < 4; ++ni) {
    int n = min(col0 + ni * 16 + ll, N - 1);
    pb[ni] = W + (size_t)n * K + hh * 8;
  }

  v8f acc[2][4] = {};
  for (int k0 = 0; k0 < K; k0 += 32) {
    Frag a[2], b[4];
    #pragma unroll
    for (int mi = 0; mi < 2; ++mi) {
      a[mi].h[0] = gload8(pa[mi]);
      a[mi].h[1] = gload8(pa[mi] + 16);
      pa[mi] += 32;
    }
    #pragma unroll
    for (int ni = 0; ni < 4; ++ni) {
      b[ni].h[0] = gload8(pb[ni]);
      b[ni].h[1] = gload8(pb[ni] + 16);
      pb[ni] += 32;
    }
    #pragma unroll
    for (int mi = 0; mi < 2; ++mi)
      #pragma unroll
      for (int ni = 0; ni < 4; ++ni)
        acc[mi][ni] = __builtin_amdgcn_wmma_f32_16x16x32_f16(
            false, a[mi].v, false, b[ni].v, (short)0, acc[mi][ni], false, false);
  }

  #pragma unroll
  for (int mi = 0; mi < 2; ++mi) {
    #pragma unroll
    for (int ni = 0; ni < 4; ++ni) {
      int col = col0 + ni * 16 + ll;
      if (col >= N) continue;
      int rbase = row0 + mi * 16 + hh * 8;
      float bv = bias ? gloadf(bias + col) : 0.f;
      #pragma unroll
      for (int v = 0; v < 8; ++v) {
        int r = rbase + v;
        if (r >= M) continue;
        float val = acc[mi][ni][v] + bv;
        if (R)   val += gloadf(R + (size_t)r * N + col);
        if (act) val = 0.5f * val * (1.f + erff(val * 0.70710678118f));
        if (C)  gstoref(C  + (size_t)r * N + col, val);
        if (Ch) gstoreh(Ch + (size_t)r * N + col, (_Float16)val);
      }
    }
  }
}

// ---------------------------------------------------------------------------
// Attention: one block per (b,h).  Whole 208x208 score tile lives in LDS
// (CDNA5: 320KB/WG).  QK^T is a single 16x16x32 WMMA per tile (K = DH = 32);
// masked softmax with wave32 shuffle reductions; P@V streams A-frags from
// LDS (ds_load_b128) against pre-transposed V.
// ---------------------------------------------------------------------------
__global__ __launch_bounds__(256) void attention_kernel(
    const _Float16* __restrict__ qh, const _Float16* __restrict__ kh,
    const _Float16* __restrict__ vt, const float* __restrict__ scale,
    _Float16* __restrict__ o) {
  extern __shared__ char smem[];
  float*    S  = (float*)smem;                               // 208 x 208 f32
  _Float16* Pm = (_Float16*)(smem + SEQP * SEQP * sizeof(float)); // 208 x 224 f16

  const int bh = blockIdx.x;
  const int h = bh % NH, b = bh / NH;
  const int lane = threadIdx.x & 31, wave = threadIdx.x >> 5;
  const int ll = lane & 15, hh = lane >> 4;
  const _Float16* qb = qh + (size_t)bh * SEQP * DHD;
  const _Float16* kb = kh + (size_t)bh * SEQP * DHD;
  const _Float16* vb = vt + (size_t)bh * DHD * SEQK;

  // ---- phase 1: S = Q K^T (13x13 tiles, one WMMA each) ----
  for (int job = wave; job < 13 * 13; job += 8) {
    int ti = job / 13, tj = job % 13;
    Frag a, bf;
    const _Float16* pa = qb + (size_t)(ti * 16 + ll) * DHD + hh * 8;
    a.h[0] = gload8(pa);  a.h[1] = gload8(pa + 16);
    const _Float16* pb = kb + (size_t)(tj * 16 + ll) * DHD + hh * 8;
    bf.h[0] = gload8(pb); bf.h[1] = gload8(pb + 16);
    v8f c = {};
    c = __builtin_amdgcn_wmma_f32_16x16x32_f16(false, a.v, false, bf.v,
                                               (short)0, c, false, false);
    int col = tj * 16 + ll, rbase = ti * 16 + hh * 8;
    #pragma unroll
    for (int v = 0; v < 8; ++v) S[(rbase + v) * SEQP + col] = c[v];
  }
  __syncthreads();

  // ---- phase 2: masked softmax per row, write P (f16, padded to 224) ----
  const float sc = scale[h];
  for (int i = wave; i < SEQP; i += 8) {
    if (i < NTOK) {
      float m = -1e30f;
      #pragma unroll
      for (int it = 0; it < 7; ++it) {
        int j = lane + it * 32;
        if (j < NTOK && j != i) m = fmaxf(m, S[i * SEQP + j] * sc);
      }
      #pragma unroll
      for (int o2 = 16; o2 > 0; o2 >>= 1) m = fmaxf(m, __shfl_xor(m, o2, 32));
      float pv[7];
      float sum = 0.f;
      #pragma unroll
      for (int it = 0; it < 7; ++it) {
        int j = lane + it * 32;
        float p = 0.f;
        if (j < NTOK && j != i) p = __expf(S[i * SEQP + j] * sc - m);
        pv[it] = p;
        sum += p;
      }
      #pragma unroll
      for (int o2 = 16; o2 > 0; o2 >>= 1) sum += __shfl_xor(sum, o2, 32);
      float inv = 1.f / sum;
      #pragma unroll
      for (int it = 0; it < 7; ++it) {
        int j = lane + it * 32;
        Pm[i * SEQK + j] = (_Float16)(pv[it] * inv);
      }
    } else {
      #pragma unroll
      for (int it = 0; it < 7; ++it) Pm[i * SEQK + lane + it * 32] = (_Float16)0.f;
    }
  }
  __syncthreads();

  // ---- phase 3: O = P @ V (13x2 tiles, K = 224 = 7 WMMA steps) ----
  for (int job = wave; job < 13 * 2; job += 8) {
    int ti = job >> 1, tj = job & 1;
    v8f c = {};
    #pragma unroll
    for (int k0 = 0; k0 < SEQK; k0 += 32) {
      Frag a, bf;
      const _Float16* pa = Pm + (size_t)(ti * 16 + ll) * SEQK + k0 + hh * 8;
      a.h[0] = *(const v8h*)pa;  a.h[1] = *(const v8h*)(pa + 16);   // LDS (ds_load_b128)
      const _Float16* pb = vb + (size_t)(tj * 16 + ll) * SEQK + k0 + hh * 8;
      bf.h[0] = gload8(pb); bf.h[1] = gload8(pb + 16);
      c = __builtin_amdgcn_wmma_f32_16x16x32_f16(false, a.v, false, bf.v,
                                                 (short)0, c, false, false);
    }
    int d = tj * 16 + ll, rbase = ti * 16 + hh * 8;
    #pragma unroll
    for (int v = 0; v < 8; ++v) {
      int i = rbase + v;
      if (i < NTOK)
        gstoreh(o + ((size_t)b * NTOK + i) * DIMD + h * DHD + d, (_Float16)c[v]);
    }
  }
}

// ---------------------------------------------------------------------------
// Host orchestration
// ---------------------------------------------------------------------------
extern "C" void kernel_launch(void* const* d_in, const int* in_sizes, int n_in,
                              void* d_out, int out_size, void* d_ws, size_t ws_size,
                              hipStream_t stream) {
  (void)in_sizes; (void)n_in; (void)out_size; (void)ws_size;
  const float* img     = (const float*)d_in[0];
  const float* patch_w = (const float*)d_in[1];
  const float* patch_b = (const float*)d_in[2];
  const float* pos_emb = (const float*)d_in[3];
  const float* cls_tok = (const float*)d_in[4];
  const float* ln1_g   = (const float*)d_in[5];
  const float* ln1_b   = (const float*)d_in[6];
  const float* qkv_w   = (const float*)d_in[7];
  const float* scale   = (const float*)d_in[8];
  const float* out_w   = (const float*)d_in[9];
  const float* out_b   = (const float*)d_in[10];
  const float* ln2_g   = (const float*)d_in[11];
  const float* ln2_b   = (const float*)d_in[12];
  const float* ff_w1   = (const float*)d_in[13];
  const float* ff_b1   = (const float*)d_in[14];
  const float* ff_w2   = (const float*)d_in[15];
  const float* ff_b2   = (const float*)d_in[16];
  const float* lnf_g   = (const float*)d_in[17];
  const float* lnf_b   = (const float*)d_in[18];
  const float* head_w  = (const float*)d_in[19];
  const float* head_b  = (const float*)d_in[20];
  float* out = (float*)d_out;

  char* ws = (char*)d_ws;
  size_t off = 0;
  auto alloc = [&](size_t bytes) -> void* {
    void* p = ws + off;
    off = (off + bytes + 255) & ~(size_t)255;
    return p;
  };

  const size_t M  = (size_t)BB * NTOK;     // 12608 token rows
  const size_t Mp = (size_t)BB * NPAT;     // 12544 patch rows

  _Float16* wPatchH = (_Float16*)alloc((size_t)DIMD * PDIM * 2);
  _Float16* wQkvH   = (_Float16*)alloc((size_t)DEPTH * 3 * DIMD * DIMD * 2);
  _Float16* wOutH   = (_Float16*)alloc((size_t)DEPTH * DIMD * DIMD * 2);
  _Float16* wFf1H   = (_Float16*)alloc((size_t)DEPTH * MLPD * DIMD * 2);
  _Float16* wFf2H   = (_Float16*)alloc((size_t)DEPTH * DIMD * MLPD * 2);
  _Float16* wHeadH  = (_Float16*)alloc((size_t)NCLS * DIMD * 2);
  _Float16* Apatch  = (_Float16*)alloc(Mp * PDIM * 2);
  float*    xpatch  = (float*)   alloc(Mp * DIMD * 4);
  float*    x       = (float*)   alloc(M * DIMD * 4);
  _Float16* hH      = (_Float16*)alloc(M * DIMD * 2);
  float*    qkv     = (float*)   alloc(M * 3 * DIMD * 4);
  _Float16* qH      = (_Float16*)alloc((size_t)BB * NH * SEQP * DHD * 2);
  _Float16* kH      = (_Float16*)alloc((size_t)BB * NH * SEQP * DHD * 2);
  _Float16* vtH     = (_Float16*)alloc((size_t)BB * NH * DHD * SEQK * 2);
  _Float16* oH      = (_Float16*)alloc(M * DIMD * 2);
  _Float16* mlpH    = (_Float16*)alloc(M * MLPD * 2);
  _Float16* clsH    = (_Float16*)alloc((size_t)BB * DIMD * 2);

  auto cvt = [&](const float* src, _Float16* dst, size_t n) {
    f32_to_f16_kernel<<<(int)((n + 255) / 256), 256, 0, stream>>>(src, dst, (int)n);
  };
  auto gemm = [&](const _Float16* A, const _Float16* W, const float* bias,
                  const float* R, float* C, _Float16* Ch,
                  int m, int n, int k, int act) {
    dim3 g((m + 127) / 128, (n + 127) / 128);
    gemm_wmma_kernel<<<g, 256, 0, stream>>>(A, W, bias, R, C, Ch, m, n, k, act);
  };

  // --- one-time (per launch) weight conversion to f16 ---
  cvt(patch_w, wPatchH, (size_t)DIMD * PDIM);
  cvt(qkv_w,   wQkvH,   (size_t)DEPTH * 3 * DIMD * DIMD);
  cvt(out_w,   wOutH,   (size_t)DEPTH * DIMD * DIMD);
  cvt(ff_w1,   wFf1H,   (size_t)DEPTH * MLPD * DIMD);
  cvt(ff_w2,   wFf2H,   (size_t)DEPTH * DIMD * MLPD);
  cvt(head_w,  wHeadH,  (size_t)NCLS * DIMD);

  // --- patch embedding ---
  {
    size_t tot = Mp * PDIM;
    build_patches_kernel<<<(int)((tot + 255) / 256), 256, 0, stream>>>(
        img, Apatch, (int)tot);
  }
  gemm(Apatch, wPatchH, patch_b, nullptr, xpatch, nullptr,
       (int)Mp, DIMD, PDIM, 0);
  {
    size_t tot = M * DIMD;
    assemble_x_kernel<<<(int)((tot + 255) / 256), 256, 0, stream>>>(
        xpatch, cls_tok, pos_emb, x, (int)tot);
  }

  const size_t attnLds = (size_t)SEQP * SEQP * sizeof(float) +
                         (size_t)SEQP * SEQK * sizeof(_Float16);

  // --- transformer layers ---
  for (int i = 0; i < DEPTH; ++i) {
    layernorm_kernel<<<(int)M, 128, 0, stream>>>(
        x, DIMD, ln1_g + i * DIMD, ln1_b + i * DIMD, hH);
    gemm(hH, wQkvH + (size_t)i * 3 * DIMD * DIMD, nullptr, nullptr,
         qkv, nullptr, (int)M, 3 * DIMD, DIMD, 0);
    {
      size_t tot = (size_t)BB * NH * SEQK * DHD;
      qkv_split_kernel<<<(int)((tot + 255) / 256), 256, 0, stream>>>(
          qkv, qH, kH, vtH, (int)tot);
    }
    attention_kernel<<<BB * NH, 256, attnLds, stream>>>(
        qH, kH, vtH, scale + i * NH, oH);
    gemm(oH, wOutH + (size_t)i * DIMD * DIMD, out_b + i * DIMD, x,
         x, nullptr, (int)M, DIMD, DIMD, 0);
    layernorm_kernel<<<(int)M, 128, 0, stream>>>(
        x, DIMD, ln2_g + i * DIMD, ln2_b + i * DIMD, hH);
    gemm(hH, wFf1H + (size_t)i * MLPD * DIMD, ff_b1 + i * MLPD, nullptr,
         nullptr, mlpH, (int)M, MLPD, DIMD, 1 /*gelu*/);
    gemm(mlpH, wFf2H + (size_t)i * DIMD * MLPD, ff_b2 + i * DIMD, x,
         x, nullptr, (int)M, DIMD, MLPD, 0);
  }

  // --- final LN on cls rows + head ---
  layernorm_kernel<<<BB, 128, 0, stream>>>(
      x, (size_t)NTOK * DIMD, lnf_g, lnf_b, clsH);
  gemm(clsH, wHeadH, head_b, nullptr, out, nullptr, BB, NCLS, DIMD, 0);
}
